// MACEModel_75402445848906
// MI455X (gfx1250) — matrix operand order repaired and checked by reading, not compiled, not run on hardware
//
#include <hip/hip_runtime.h>

#define NE 16384
#define NN 2048
#define NG 64

typedef __attribute__((ext_vector_type(16))) __bf16          v16bf;
typedef __attribute__((ext_vector_type(16))) unsigned short  v16us;
typedef __attribute__((ext_vector_type(8)))  unsigned short  v8us;
typedef __attribute__((ext_vector_type(8)))  float           v8f;

// ======================= compile-time real Clebsch-Gordan =======================
struct cpx { double re; double im; };
constexpr cpx cmul(cpx a, cpx b){ return cpx{a.re*b.re - a.im*b.im, a.re*b.im + a.im*b.re}; }
constexpr cpx cconj(cpx a){ return cpx{a.re, -a.im}; }

constexpr double cfact(int n){ double r=1.0; for(int i=2;i<=n;++i) r*=(double)i; return r; }

constexpr double csqrt_(double x){
  if(x<=0.0) return 0.0;
  double g = x>=1.0 ? x : 1.0, prev = 0.0;
  for(int i=0;i<64 && g!=prev;++i){ prev=g; g=0.5*(g+x/g); }
  return g;
}

constexpr double cg_complex(int l1,int l2,int l3,int m1,int m2,int m3){
  if(m1+m2!=m3) return 0.0;
  int lmin = l1>l2 ? l1-l2 : l2-l1;
  if(l3<lmin || l3>l1+l2) return 0.0;
  double pre = csqrt_((2.0*l3+1.0)*cfact(l1+l2-l3)*cfact(l1-l2+l3)*cfact(-l1+l2+l3)/cfact(l1+l2+l3+1));
  pre *= csqrt_(cfact(l3+m3)*cfact(l3-m3)*cfact(l1+m1)*cfact(l1-m1)*cfact(l2+m2)*cfact(l2-m2));
  int kmin = 0;
  if(l2-l3-m1>kmin) kmin=l2-l3-m1;
  if(l1-l3+m2>kmin) kmin=l1-l3+m2;
  int kmax = l1+l2-l3;
  if(l1-m1<kmax) kmax=l1-m1;
  if(l2+m2<kmax) kmax=l2+m2;
  double s=0.0;
  for(int k=kmin;k<=kmax;++k){
    double d = cfact(k)*cfact(l1+l2-l3-k)*cfact(l1-m1-k)*cfact(l2+m2-k)*cfact(l3-l2+m1+k)*cfact(l3-l1-m2+k);
    s += ((k&1)? -1.0 : 1.0)/d;
  }
  return pre*s;
}

constexpr cpx umat(int l,int r,int c){
  double s2 = 1.0/csqrt_(2.0);
  if(r==l) return (c==l)? cpx{1.0,0.0} : cpx{0.0,0.0};
  int m = r-l;
  if(m>0){
    if(c==l-m) return cpx{s2,0.0};
    if(c==l+m) return cpx{ (m&1)? -s2 : s2, 0.0};
  } else {
    int mm=-m;
    if(c==l-mm) return cpx{0.0, s2};
    if(c==l+mm) return cpx{0.0, (mm&1)? s2 : -s2};
  }
  return cpx{0.0,0.0};
}

struct CGT { float v[5][5][5]; };

constexpr CGT make_cg(int l1,int l2,int l3){
  CGT t{};
  const int d1=2*l1+1, d2=2*l2+1, d3=2*l3+1;
  cpx U1[5][5]{}; cpx U2[5][5]{}; cpx U3[5][5]{};
  for(int r=0;r<d1;++r) for(int c=0;c<d1;++c) U1[r][c]=umat(l1,r,c);
  for(int r=0;r<d2;++r) for(int c=0;c<d2;++c) U2[r][c]=umat(l2,r,c);
  for(int r=0;r<d3;++r) for(int c=0;c<d3;++c) U3[r][c]=umat(l3,r,c);
  double cg[5][5]{};
  for(int a=0;a<d1;++a) for(int b=0;b<d2;++b){
    int m3=(a-l1)+(b-l2);
    cg[a][b] = (m3<-l3||m3>l3)? 0.0 : cg_complex(l1,l2,l3,a-l1,b-l2,m3);
  }
  for(int i=0;i<d1;++i) for(int j=0;j<d2;++j) for(int k=0;k<d3;++k){
    double re=0.0;
    for(int a=0;a<d1;++a) for(int b=0;b<d2;++b){
      double g=cg[a][b]; if(g==0.0) continue;
      int c=(a-l1)+(b-l2)+l3;
      cpx p = cmul(cconj(U1[i][a]), cconj(U2[j][b]));
      p = cmul(p, U3[k][c]);
      re += p.re*g;
    }
    t.v[i][j][k]=(float)re;
  }
  return t;
}

constexpr CGT CGc0  = make_cg(0,0,0);
constexpr CGT CGc1  = make_cg(0,1,1);
constexpr CGT CGc2  = make_cg(0,2,2);
constexpr CGT CGc3  = make_cg(1,0,1);
constexpr CGT CGc4  = make_cg(1,1,0);
constexpr CGT CGc5  = make_cg(1,1,2);
constexpr CGT CGc6  = make_cg(1,2,1);
constexpr CGT CGc7  = make_cg(2,0,2);
constexpr CGT CGc8  = make_cg(2,1,1);
constexpr CGT CGc9  = make_cg(2,2,0);
constexpr CGT CGc10 = make_cg(2,2,2);

__device__ const CGT CGALL[11] = {CGc0,CGc1,CGc2,CGc3,CGc4,CGc5,CGc6,CGc7,CGc8,CGc9,CGc10};
__device__ const int PATH_L[11][3] = {{0,0,0},{0,1,1},{0,2,2},{1,0,1},{1,1,0},{1,1,2},
                                      {1,2,1},{2,0,2},{2,1,1},{2,2,0},{2,2,2}};
__device__ const int LOFF[3] = {0,1,4};

constexpr float inv_sqrtc(double x){ return (float)(1.0/csqrt_(x)); }
__device__ const float SCALE_FIRST[3]  = { inv_sqrtc(16.0), inv_sqrtc(16.0), inv_sqrtc(16.0) };
__device__ const float SCALE_FULL[11]  = {
  inv_sqrtc(48.0), inv_sqrtc(64.0), inv_sqrtc(64.0), inv_sqrtc(64.0),
  inv_sqrtc(48.0), inv_sqrtc(64.0), inv_sqrtc(64.0), inv_sqrtc(64.0),
  inv_sqrtc(64.0), inv_sqrtc(48.0), inv_sqrtc(64.0) };

// ======================= helpers =======================
__device__ __forceinline__ unsigned short f2bf(float f){
  unsigned int u = __float_as_uint(f);
  u += 0x7FFFu + ((u>>16)&1u);
  return (unsigned short)(u>>16);
}

__global__ void k_zero(float* __restrict__ p, int n){
  int i = blockIdx.x*blockDim.x + threadIdx.x;
  if(i<n) p[i]=0.0f;
}

// Convert w2 (f32, [256][Ncols]) into bf16 *pre-swizzled WMMA B-fragment* layout:
//   dst[((nt*8 + ks)*32 + lane)*16 + e] = bf16( w2[(ks*32 + 16*(e>>3) + 8*(lane>>4) + (e&7))*Ncols + nt*16 + (lane&15)] )
// so each lane's per-k-step fragment is one aligned 32-byte run.
__global__ void k_cvt_w2_frag(const float* __restrict__ s, unsigned short* __restrict__ d,
                              int Ncols, int n){
  int i = blockIdx.x*blockDim.x + threadIdx.x;
  if(i>=n) return;
  int e    =  i      & 15;
  int lane = (i>>4)  & 31;
  int ks   = (i>>9)  & 7;
  int nt   =  i>>12;
  int krow = ks*32 + ((e>>3)<<4) + ((lane>>4)<<3) + (e&7);
  int col  = nt*16 + (lane&15);
  d[i] = f2bf(s[krow*Ncols + col]);
}

// ======================= edge basis: SH + Bessel*envelope =======================
__global__ void k_edge_basis(const float* __restrict__ pos, const int* __restrict__ ei,
                             float* __restrict__ shb, float* __restrict__ feats){
  int e = blockIdx.x*blockDim.x + threadIdx.x;
  if(e>=NE) return;
  int s = ei[e], d = ei[NE+e];
  float vx = pos[s*3+0]-pos[d*3+0];
  float vy = pos[s*3+1]-pos[d*3+1];
  float vz = pos[s*3+2]-pos[d*3+2];
  float r = sqrtf(vx*vx+vy*vy+vz*vz+1e-12f);
  float inv = 1.0f/r;
  float x=vx*inv, y=vy*inv, z=vz*inv;
  const float c3=1.7320508075688772f, c15=3.872983346207417f, c5h=1.118033988749895f;
  float* o = shb + e*9;
  o[0]=1.0f; o[1]=c3*y; o[2]=c3*z; o[3]=c3*x;
  o[4]=c15*x*y; o[5]=c15*y*z; o[6]=c5h*(3.0f*z*z-1.0f);
  o[7]=c15*x*z; o[8]=0.5f*c15*(x*x-y*y);
  float xr = r*0.1f;
  float env = 0.0f;
  if(xr<1.0f){
    float x5=xr*xr*xr*xr*xr;
    env = 1.0f - 21.0f*x5 + 35.0f*x5*xr - 15.0f*x5*xr*xr;
  }
  const float pref = 0.4472135954999579f; // sqrt(2/10)
  float* f = feats + e*8;
  float w = 3.14159265358979323846f * r * 0.1f;
  #pragma unroll
  for(int n=1;n<=8;++n) f[n-1] = pref * sinf(w*(float)n) * inv * env;
}

__global__ void k_init_h9(const int* __restrict__ atoms, const float* __restrict__ emb,
                          float* __restrict__ h9){
  int i = blockIdx.x*blockDim.x + threadIdx.x;
  if(i>=NN*144) return;
  int n=i/144, rem=i%144, c=rem/9, k=rem%9;
  h9[i] = (k==0)? emb[atoms[n]*16 + c] : 0.0f;
}

// ======================= radial MLP layer 1 (tiny, VALU) =======================
__global__ void k_mlp1(const float* __restrict__ feats, const float* __restrict__ w1,
                       const float* __restrict__ b1, unsigned short* __restrict__ hid){
  __shared__ float f8[8];
  int e = blockIdx.x;
  int h = threadIdx.x;
  if(h<8) f8[h] = feats[e*8+h];
  __syncthreads();
  float a = b1[h];
  #pragma unroll
  for(int i=0;i<8;++i) a += f8[i]*w1[i*256+h];
  a = a>0.0f ? a : 0.0f;
  hid[e*256+h] = f2bf(a);
}

// ======================= radial MLP layer 2: WMMA bf16 GEMM =======================
// 4 waves / block, each wave owns one 16-wide column tile.
// wz[rowChunkLocal, col] = hidden[rowBase+row, :256] @ w2[:256, col] + b2[col]
__global__ void __launch_bounds__(128)
k_gemm2(const unsigned short* __restrict__ A,     // hidden bf16 [NE][256]
        const unsigned short* __restrict__ Bfrag, // w2 bf16, fragment-swizzled
        const float* __restrict__ bias,
        float* __restrict__ Cbuf,                 // [2048][Ncols]
        int Ncols, int rowBase){
  const int lane = threadIdx.x & 31;
  const int wave = threadIdx.x >> 5;
  const int g    = lane >> 4;
  const int ln   = lane & 15;
  const int mT   = blockIdx.x << 4;                 // chunk-local row tile
  const int ntile= (blockIdx.y<<2) + wave;          // column tile index
  const int nT   = ntile << 4;
  const int rowA = rowBase + mT + ln;

  const unsigned short* Arow = A + (size_t)rowA*256;
  const v16us* Bf = (const v16us*)Bfrag;            // 32B fragments

  v8f acc = {0.0f,0.0f,0.0f,0.0f,0.0f,0.0f,0.0f,0.0f};
  #pragma unroll 2
  for(int ks=0;ks<8;++ks){
    if(ks+1<8) __builtin_prefetch(&Bf[(ntile*8 + ks+1)*32 + lane], 0, 1);
    // A layout (16-bit 16x32): lane M=L%16; K = 16*(e>>3) + 8*g + (e&7)
    // -> two contiguous 16-byte runs per lane
    v8us a0 = *(const v8us*)(Arow + ks*32 + (g<<3));
    v8us a1 = *(const v8us*)(Arow + ks*32 + 16 + (g<<3));
    v16us au = __builtin_shufflevector(a0, a1, 0,1,2,3,4,5,6,7,8,9,10,11,12,13,14,15);
    // B fragment: one aligned 32-byte vector load per lane
    v16us bu = Bf[(ntile*8 + ks)*32 + lane];
    v16bf a = __builtin_bit_cast(v16bf, au);
    v16bf b = __builtin_bit_cast(v16bf, bu);
    acc = __builtin_amdgcn_wmma_f32_16x16x32_bf16(false, a, false, b, (short)0, acc, false, false);
  }
  const int col = nT + ln;
  const float bv = bias[col];
  #pragma unroll
  for(int r=0;r<8;++r){
    int m = mT + r + (g<<3);   // C/D layout: VGPR r -> M = r + 8*g
    Cbuf[m*Ncols + col] = acc[r] + bv;
  }
}

// ======================= tensor-product conv (per-edge) =======================
__global__ void k_conv(const float* __restrict__ h9, const int* __restrict__ ei,
                       const float* __restrict__ shb, const float* __restrict__ wz,
                       float* __restrict__ up, int base, int Ncols, int full){
  __shared__ float mbuf[8][16][36];
  const int el = threadIdx.x >> 4;   // 8 edges per block
  const int ln = threadIdx.x & 15;
  const int e  = base + (blockIdx.x<<3) + el;
  const int npaths = full ? 11 : 3;
  // phase 1: ln = u (channel) -> m[u][k] per path
  {
    const int src = ei[e];
    float hs[9], s9[9];
    #pragma unroll
    for(int i=0;i<9;++i) hs[i] = h9[src*144 + ln*9 + i];
    #pragma unroll
    for(int j=0;j<9;++j) s9[j] = shb[e*9 + j];
    int kslot=0;
    for(int p=0;p<npaths;++p){
      int l1=PATH_L[p][0], l2=PATH_L[p][1], l3=PATH_L[p][2];
      int o1=LOFF[l1], o2=LOFF[l2];
      int n1=2*l1+1, n2=2*l2+1, n3=2*l3+1;
      for(int k=0;k<n3;++k){
        float a=0.0f;
        for(int i=0;i<n1;++i){
          float hv = hs[o1+i];
          for(int j=0;j<n2;++j)
            a += hv * s9[o2+j] * CGALL[p].v[i][j][k];
        }
        mbuf[el][ln][kslot+k]=a;
      }
      kslot += n3;
    }
  }
  __syncthreads();
  // phase 2: ln = v (output channel): out[v][k] = sum_u m[u][k]*W[u][v]
  float msg[9];
  #pragma unroll
  for(int t=0;t<9;++t) msg[t]=0.0f;
  const int erow = (blockIdx.x<<3) + el;      // chunk-local wz row
  int kslot=0;
  for(int p=0;p<npaths;++p){
    int l3=PATH_L[p][2], o3=LOFF[l3], n3=2*l3+1;
    float sc = full ? SCALE_FULL[p] : SCALE_FIRST[p];
    const float* wr = wz + (size_t)erow*Ncols + p*256 + ln;  // W[u][v] = wz[p*256 + u*16 + v]
    for(int k=0;k<n3;++k){
      float a=0.0f;
      #pragma unroll
      for(int u=0;u<16;++u) a += mbuf[el][u][kslot+k]*wr[u*16];
      msg[o3+k] += a*sc;
    }
    kslot += n3;
  }
  const int dst = ei[NE + e];
  #pragma unroll
  for(int t=0;t<9;++t) atomicAdd(&up[dst*144 + ln*9 + t], msg[t]);
}

// ======================= node-wise CG products =======================
__device__ __forceinline__ void cg_prod_dev(const float* X, const float* Y, float* O){
  #pragma unroll
  for(int t=0;t<9;++t) O[t]=0.0f;
  for(int p=0;p<11;++p){
    int l1=PATH_L[p][0], l2=PATH_L[p][1], l3=PATH_L[p][2];
    int o1=LOFF[l1], o2=LOFF[l2], o3=LOFF[l3];
    int n1=2*l1+1, n2=2*l2+1, n3=2*l3+1;
    for(int k=0;k<n3;++k){
      float a=0.0f;
      for(int i=0;i<n1;++i){
        float xv = X[o1+i];
        for(int j=0;j<n2;++j) a += xv*Y[o2+j]*CGALL[p].v[i][j][k];
      }
      O[o3+k]+=a;
    }
  }
  O[0] *= 0.5773502691896258f;  // 1/sqrt(NP_FULL[0]=3)
  #pragma unroll
  for(int t=1;t<9;++t) O[t]*=0.5f;  // 1/sqrt(4)
}

__global__ void k_prod(const float* __restrict__ up, float* __restrict__ h9,
                       const float* __restrict__ Wp){
  __shared__ float buf[8][16][27];
  const int nl = threadIdx.x>>4, ln = threadIdx.x&15;
  const int n  = (blockIdx.x<<3)+nl;
  {
    float A[9], B2[9], B3[9];
    #pragma unroll
    for(int k=0;k<9;++k) A[k]=up[n*144+ln*9+k];
    cg_prod_dev(A,A,B2);
    cg_prod_dev(B2,A,B3);
    #pragma unroll
    for(int k=0;k<9;++k){ buf[nl][ln][k]=A[k]; buf[nl][ln][9+k]=B2[k]; buf[nl][ln][18+k]=B3[k]; }
  }
  __syncthreads();
  const float inv3c = 0.14433756729740643f;  // 1/sqrt(3*C)
  float out[9];
  for(int l=0;l<3;++l){
    int o=LOFF[l], nk=2*l+1;
    for(int k=0;k<nk;++k){
      float acc=0.0f;
      #pragma unroll
      for(int c=0;c<16;++c){
        float w0 = Wp[(((0*3+l)*16+c)<<4)+ln];
        float w1 = Wp[(((1*3+l)*16+c)<<4)+ln];
        float w2 = Wp[(((2*3+l)*16+c)<<4)+ln];
        acc += buf[nl][c][o+k]*w0 + buf[nl][c][9+o+k]*w1 + buf[nl][c][18+o+k]*w2;
      }
      out[o+k] = acc*inv3c + h9[n*144+ln*9+o+k];
    }
  }
  #pragma unroll
  for(int t=0;t<9;++t) h9[n*144+ln*9+t]=out[t];
}

// ======================= pooling + prediction MLP =======================
__global__ void k_pool(const float* __restrict__ h9, const int* __restrict__ batch,
                       float* __restrict__ pooled){
  int i=blockIdx.x*blockDim.x+threadIdx.x;
  if(i>=NN*16) return;
  int n=i>>4, c=i&15;
  atomicAdd(&pooled[batch[n]*16+c], h9[n*144+c*9]);
}

__global__ void k_pred(const float* __restrict__ pooled, const float* __restrict__ w1,
                       const float* __restrict__ b1, const float* __restrict__ w2,
                       const float* __restrict__ b2, float* __restrict__ out){
  int g=threadIdx.x;
  if(g>=NG) return;
  float p[16];
  #pragma unroll
  for(int c=0;c<16;++c) p[c]=pooled[g*16+c];
  float o=b2[0];
  for(int d=0;d<16;++d){
    float t=b1[d];
    #pragma unroll
    for(int c=0;c<16;++c) t += p[c]*w1[c*16+d];
    t = t>0.0f? t:0.0f;
    o += t*w2[d];
  }
  out[g]=o;
}

// ======================= driver =======================
extern "C" void kernel_launch(void* const* d_in, const int* in_sizes, int n_in,
                              void* d_out, int out_size, void* d_ws, size_t ws_size,
                              hipStream_t stream){
  const int*   atoms = (const int*)d_in[0];
  const float* pos   = (const float*)d_in[1];
  const int*   ei    = (const int*)d_in[2];
  const int*   batch = (const int*)d_in[3];
  const float* emb   = (const float*)d_in[4];
  const float* c0w1=(const float*)d_in[5];  const float* c0b1=(const float*)d_in[6];
  const float* c0w2=(const float*)d_in[7];  const float* c0b2=(const float*)d_in[8];
  const float* c1w1=(const float*)d_in[9];  const float* c1b1=(const float*)d_in[10];
  const float* c1w2=(const float*)d_in[11]; const float* c1b2=(const float*)d_in[12];
  const float* p0w =(const float*)d_in[13]; const float* p1w =(const float*)d_in[14];
  const float* pw1 =(const float*)d_in[15]; const float* pb1 =(const float*)d_in[16];
  const float* pw2 =(const float*)d_in[17]; const float* pb2 =(const float*)d_in[18];
  float* outp=(float*)d_out;

  char* w=(char*)d_ws;
  auto au=[](size_t x){ return (x+255)&~(size_t)255; };
  size_t off=0;
  float*          shb   =(float*)(w+off);          off=au(off+(size_t)NE*9*4);
  float*          feats =(float*)(w+off);          off=au(off+(size_t)NE*8*4);
  unsigned short* hid   =(unsigned short*)(w+off); off=au(off+(size_t)NE*256*2);
  unsigned short* w2b   =(unsigned short*)(w+off); off=au(off+(size_t)256*2816*2);
  float*          wz    =(float*)(w+off);          off=au(off+(size_t)2048*2816*4);
  float*          up    =(float*)(w+off);          off=au(off+(size_t)NN*144*4);
  float*          h9    =(float*)(w+off);          off=au(off+(size_t)NN*144*4);
  float*          pooled=(float*)(w+off);          off=au(off+(size_t)NG*16*4);

  k_edge_basis<<<NE/256,256,0,stream>>>(pos,ei,shb,feats);
  k_init_h9<<<(NN*144+255)/256,256,0,stream>>>(atoms,emb,h9);

  // ---- conv0 (3 paths, Ncols=768) + prod0 ----
  k_mlp1<<<NE,256,0,stream>>>(feats,c0w1,c0b1,hid);
  k_cvt_w2_frag<<<(256*768+255)/256,256,0,stream>>>(c0w2,w2b,768,256*768);
  k_zero<<<(NN*144+255)/256,256,0,stream>>>(up,NN*144);
  for(int c=0;c<8;++c){
    k_gemm2<<<dim3(2048/16, 768/64),128,0,stream>>>(hid,w2b,c0b2,wz,768,c*2048);
    k_conv<<<2048/8,128,0,stream>>>(h9,ei,shb,wz,up,c*2048,768,0);
  }
  k_prod<<<NN/8,128,0,stream>>>(up,h9,p0w);

  // ---- conv1 (11 paths, Ncols=2816) + prod1 ----
  k_mlp1<<<NE,256,0,stream>>>(feats,c1w1,c1b1,hid);
  k_cvt_w2_frag<<<(256*2816+255)/256,256,0,stream>>>(c1w2,w2b,2816,256*2816);
  k_zero<<<(NN*144+255)/256,256,0,stream>>>(up,NN*144);
  for(int c=0;c<8;++c){
    k_gemm2<<<dim3(2048/16, 2816/64),128,0,stream>>>(hid,w2b,c1b2,wz,2816,c*2048);
    k_conv<<<2048/8,128,0,stream>>>(h9,ei,shb,wz,up,c*2048,2816,1);
  }
  k_prod<<<NN/8,128,0,stream>>>(up,h9,p1w);

  // ---- pooling + prediction ----
  k_zero<<<1,256,0,stream>>>(pooled,NG*16);
  k_pool<<<(NN*16+255)/256,256,0,stream>>>(h9,batch,pooled);
  k_pred<<<1,64,0,stream>>>(pooled,pw1,pb1,pw2,pb2,outp);

  (void)in_sizes;(void)n_in;(void)out_size;(void)ws_size;
}
